// AttShare_25967372271655
// MI455X (gfx1250) — matrix operand order repaired and checked by reading, not compiled, not run on hardware
//
#include <hip/hip_runtime.h>

// CDNA5 / gfx1250 WMMA types (wave32)
typedef __attribute__((ext_vector_type(16))) _Float16 v16h;
typedef __attribute__((ext_vector_type(8)))  _Float16 v8h;
typedef __attribute__((ext_vector_type(8)))  float    v8f;

// ---------------------------------------------------------------------------
// CDNA5 fragment layouts (CDNA5 ISA §7.12.2), wave32:
//  A 16x32 f16 : lane = 16*half + m ; elem pair (2v,2v+1) <- K = (v>=4?16:0)+8*half+2*(v%4)
//                => elems 0..7  = A[m][k0+ 8*half .. +7]   (contiguous K run)
//                   elems 8..15 = A[m][k0+16+8*half .. +7] (contiguous K run)
//  B 32x16 f16 : lane = 16*half + n ; elem pair (2v,2v+1) <- K = (v>=4?16:0)+8*half+2*(v%4)
//                => elems 0..7 = B[k0+8*half .. +7][n], 8..15 = B[k0+16+8*half .. +7][n] (fixed n)
//  C/D 16x16 f32: lane = 16*half + n ; VGPR v <-> M = v + 8*half
// ---------------------------------------------------------------------------

static __device__ __forceinline__ v16h frag_a_f32_rm(const float* __restrict__ A,
                                                     int lda, int m0, int k0) {
  const int lane = threadIdx.x & 31;
  const int m    = m0 + (lane & 15);
  const int half = lane >> 4;
  v16h f;
#pragma unroll
  for (int v = 0; v < 8; ++v) {
    const int kk = k0 + ((v >> 2) << 4) + (half << 3) + ((v & 3) << 1);
    f[2 * v]     = (_Float16)A[(size_t)m * lda + kk];
    f[2 * v + 1] = (_Float16)A[(size_t)m * lda + kk + 1];
  }
  return f;
}

static __device__ __forceinline__ v16h frag_b_f32_rm(const float* __restrict__ B,
                                                     int ldb, int k0, int n0) {
  const int lane = threadIdx.x & 31;
  const int n    = n0 + (lane & 15);
  const int half = lane >> 4;
  v16h f;
#pragma unroll
  for (int v = 0; v < 8; ++v) {
    const int kk = k0 + ((v >> 2) << 4) + (half << 3) + ((v & 3) << 1);
    f[2 * v]     = (_Float16)B[(size_t)kk * ldb + n];
    f[2 * v + 1] = (_Float16)B[(size_t)(kk + 1) * ldb + n];
  }
  return f;
}

static __device__ __forceinline__ v16h frag_a_f16_tr(const _Float16* __restrict__ Q,
                                                     int ldq, int m0, int k0) {
  // A[m][k] = Q[k][m]  (A = Q^T)
  const int lane = threadIdx.x & 31;
  const int m    = m0 + (lane & 15);
  const int half = lane >> 4;
  v16h f;
#pragma unroll
  for (int v = 0; v < 8; ++v) {
    const int kk = k0 + ((v >> 2) << 4) + (half << 3) + ((v & 3) << 1);
    f[2 * v]     = Q[(size_t)kk * ldq + m];
    f[2 * v + 1] = Q[(size_t)(kk + 1) * ldq + m];
  }
  return f;
}

static __device__ __forceinline__ v16h frag_b_f16_rm(const _Float16* __restrict__ B,
                                                     int ldb, int k0, int n0) {
  const int lane = threadIdx.x & 31;
  const int n    = n0 + (lane & 15);
  const int half = lane >> 4;
  v16h f;
#pragma unroll
  for (int v = 0; v < 8; ++v) {
    const int kk = k0 + ((v >> 2) << 4) + (half << 3) + ((v & 3) << 1);
    f[2 * v]     = B[(size_t)kk * ldb + n];
    f[2 * v + 1] = B[(size_t)(kk + 1) * ldb + n];
  }
  return f;
}

// ---------------------------------------------------------------------------
// K1: Y[o][n] = f16( sum_c W[o][c]*X[c][n] + bias[o] )   (q/k/v projections)
// One wave computes a 16x64 strip: A(W) fragment loaded once per K-step.
// ---------------------------------------------------------------------------
__global__ __launch_bounds__(32) void proj_gemm_f16out(
    const float* __restrict__ W, const float* __restrict__ bias,
    const float* __restrict__ X, _Float16* __restrict__ Y,
    int Cin, int N) {
  const int n0 = blockIdx.x * 64;
  const int m0 = blockIdx.y * 16;
  v8f acc[4] = {{}, {}, {}, {}};
  for (int k0 = 0; k0 < Cin; k0 += 32) {
    const v16h a = frag_a_f32_rm(W, Cin, m0, k0);
#pragma unroll
    for (int nt = 0; nt < 4; ++nt) {
      const v16h b = frag_b_f32_rm(X, N, k0, n0 + nt * 16);
      acc[nt] = __builtin_amdgcn_wmma_f32_16x16x32_f16(false, a, false, b,
                                                       (short)0, acc[nt], false, false);
    }
  }
  const int lane = threadIdx.x & 31;
  const int half = lane >> 4;
#pragma unroll
  for (int nt = 0; nt < 4; ++nt) {
    const int n = n0 + nt * 16 + (lane & 15);
#pragma unroll
    for (int v = 0; v < 8; ++v) {
      const int m = m0 + v + (half << 3);
      Y[(size_t)m * N + n] = (_Float16)(acc[nt][v] + bias[m]);
    }
  }
}

// ---------------------------------------------------------------------------
// K2: S[i][j] = sum_c Q[c][i]*K[c][j]   (logits, f16 out)
// The reference's gating bias is constant per row i -> exactly droppable
// under softmax(axis=j). One wave = 16x64 strip (transposed-A loaded once).
// ---------------------------------------------------------------------------
__global__ __launch_bounds__(32) void attn_logits(
    const _Float16* __restrict__ Q, const _Float16* __restrict__ Km,
    _Float16* __restrict__ S, int Cq, int N) {
  const int n0 = blockIdx.x * 64;  // j
  const int m0 = blockIdx.y * 16;  // i
  v8f acc[4] = {{}, {}, {}, {}};
  for (int k0 = 0; k0 < Cq; k0 += 32) {
    const v16h a = frag_a_f16_tr(Q, N, m0, k0);
#pragma unroll
    for (int nt = 0; nt < 4; ++nt) {
      const v16h b = frag_b_f16_rm(Km, N, k0, n0 + nt * 16);
      acc[nt] = __builtin_amdgcn_wmma_f32_16x16x32_f16(false, a, false, b,
                                                       (short)0, acc[nt], false, false);
    }
  }
  const int lane = threadIdx.x & 31;
  const int half = lane >> 4;
#pragma unroll
  for (int nt = 0; nt < 4; ++nt) {
    const int n = n0 + nt * 16 + (lane & 15);
#pragma unroll
    for (int v = 0; v < 8; ++v) {
      const int m = m0 + v + (half << 3);
      S[(size_t)m * N + n] = (_Float16)acc[nt][v];
    }
  }
}

// ---------------------------------------------------------------------------
// K3: in-place row softmax over N f16 logits (f32 math)
// ---------------------------------------------------------------------------
__global__ __launch_bounds__(256) void softmax_rows(_Float16* __restrict__ S, int N) {
  _Float16* p = S + (size_t)blockIdx.x * N;
  __shared__ float red[256];
  const int t = threadIdx.x;

  float mx = -3.0e38f;
  for (int j = t; j < N; j += 256) mx = fmaxf(mx, (float)p[j]);
  red[t] = mx;
  __syncthreads();
  for (int s = 128; s > 0; s >>= 1) {
    if (t < s) red[t] = fmaxf(red[t], red[t + s]);
    __syncthreads();
  }
  mx = red[0];
  __syncthreads();

  float sum = 0.0f;
  for (int j = t; j < N; j += 256) sum += expf((float)p[j] - mx);
  red[t] = sum;
  __syncthreads();
  for (int s = 128; s > 0; s >>= 1) {
    if (t < s) red[t] += red[t + s];
    __syncthreads();
  }
  const float inv = 1.0f / red[0];

  for (int j = t; j < N; j += 256) p[j] = (_Float16)(expf((float)p[j] - mx) * inv);
}

// ---------------------------------------------------------------------------
// K4: Out[c][i] = gamma * sum_j V[c][j]*P[i][j] + X0[c][i]
// LDS-staged 64x64 tile, 4 waves, TK=64 (8 WMMAs per barrier round).
// Staging uses the CDNA5 async direct global->LDS path:
//   global_load_async_to_lds_b128  (ASYNCcnt)  +  s_wait_asynccnt 0
// B^T orientation is free (B[k][n] = P[n][k] -> coalesced row reads of P).
// Padded LDS rows (stride 72 halves = 36 dwords) keep ds_load_b128 fragment
// reads bank-conflict-free (36*m mod 64 distinct for m=0..15).
// ---------------------------------------------------------------------------
#define OT_TM 64
#define OT_TN 64
#define OT_TK 64
#define OT_PAD 8  // pad rows by 8 halves (16B) -> stride 72 halves

// direct global->LDS 16B async copy; ldsoff is the wave-relative LDS byte
// offset (low 32 bits of the generic address of a __shared__ object)
#define ASYNC_CP_B128(ldsoff, gptr)                              \
  asm volatile("global_load_async_to_lds_b128 %0, %1, off"       \
               :: "v"(ldsoff), "v"(gptr) : "memory")

static __device__ __forceinline__ unsigned lds_off(const void* p) {
  return (unsigned)(size_t)p;  // generic LDS addr[31:0] == LDS byte offset
}

__global__ __launch_bounds__(128) void out_gemm_lds(
    const _Float16* __restrict__ V, const _Float16* __restrict__ P,
    const float* __restrict__ X0, const float* __restrict__ gamma,
    float* __restrict__ Out, int N) {
  __shared__ alignas(16) _Float16 As[OT_TM][OT_TK + OT_PAD];  // V tile (rows = c)
  __shared__ alignas(16) _Float16 Bs[OT_TN][OT_TK + OT_PAD];  // P tile (rows = i, Bt layout)

  const int tid  = threadIdx.x;
  const int wave = tid >> 5;     // 0..3 -> m-tile
  const int lane = tid & 31;
  const int half = lane >> 4;
  const int lmn  = lane & 15;

  const int nBase = blockIdx.x * OT_TN;  // i
  const int mBase = blockIdx.y * OT_TM;  // c

  v8f acc[4] = {{}, {}, {}, {}};

  for (int k0 = 0; k0 < N; k0 += OT_TK) {
    if (k0) __syncthreads();  // previous round's LDS readers done

    // ---- async stage V and P tiles: 64 rows x 8 chunks(16B) each matrix ----
    // 512 chunks/matrix, 128 threads -> 4 chunks per thread per matrix
#pragma unroll
    for (int t = 0; t < 4; ++t) {
      const int idx = tid + t * 128;
      const int r = idx >> 3, c8 = (idx & 7) * 8;
      ASYNC_CP_B128(lds_off(&As[r][c8]), V + (size_t)(mBase + r) * N + k0 + c8);
      ASYNC_CP_B128(lds_off(&Bs[r][c8]), P + (size_t)(nBase + r) * N + k0 + c8);
    }
    if (k0 + OT_TK < N) {  // uniform: prefetch next round (global_prefetch_b8)
      __builtin_prefetch(V + (size_t)(mBase + (tid >> 3)) * N + k0 + OT_TK, 0, 3);
      __builtin_prefetch(P + (size_t)(nBase + (tid >> 3)) * N + k0 + OT_TK, 0, 3);
    }
    asm volatile("s_wait_asynccnt 0x0" ::: "memory");
    __syncthreads();

    // ---- two K-subtiles per round: 8 WMMAs per wave per barrier pair ----
#pragma unroll
    for (int ks = 0; ks < OT_TK; ks += 32) {
      const int am = wave * 16 + lmn;
      const v8h alo = *(const v8h*)(&As[am][ks + half * 8]);
      const v8h ahi = *(const v8h*)(&As[am][ks + 16 + half * 8]);
      const v16h a = __builtin_shufflevector(alo, ahi, 0, 1, 2, 3, 4, 5, 6, 7,
                                             8, 9, 10, 11, 12, 13, 14, 15);
#pragma unroll
      for (int nt = 0; nt < 4; ++nt) {
        const int bn = nt * 16 + lmn;
        const v8h blo = *(const v8h*)(&Bs[bn][ks + half * 8]);
        const v8h bhi = *(const v8h*)(&Bs[bn][ks + 16 + half * 8]);
        const v16h b = __builtin_shufflevector(blo, bhi, 0, 1, 2, 3, 4, 5, 6, 7,
                                               8, 9, 10, 11, 12, 13, 14, 15);
        acc[nt] = __builtin_amdgcn_wmma_f32_16x16x32_f16(false, a, false, b,
                                                         (short)0, acc[nt], false, false);
      }
    }
  }

  const float g = gamma[0];
#pragma unroll
  for (int nt = 0; nt < 4; ++nt) {
    const int n = nBase + nt * 16 + lmn;
#pragma unroll
    for (int v = 0; v < 8; ++v) {
      const int m = mBase + wave * 16 + v + (half << 3);
      Out[(size_t)m * N + n] = g * acc[nt][v] + X0[(size_t)m * N + n];
    }
  }
}

// ---------------------------------------------------------------------------
// Host orchestration. Workspace (f16), reused across the 4 (b,s) passes:
//   q16[64*4096] k16[64*4096] v16[256*4096] P[4096*4096]  -> ~36.7 MB
// ---------------------------------------------------------------------------
extern "C" void kernel_launch(void* const* d_in, const int* in_sizes, int n_in,
                              void* d_out, int out_size, void* d_ws, size_t ws_size,
                              hipStream_t stream) {
  (void)in_sizes; (void)n_in; (void)out_size; (void)ws_size;

  constexpr int B = 2, C = 256, Cq = 64, N = 4096;

  const float* inp[2] = { (const float*)d_in[0], (const float*)d_in[1] };
  const float* gamma  = (const float*)d_in[22];

  _Float16* q16 = (_Float16*)d_ws;
  _Float16* k16 = q16 + (size_t)Cq * N;
  _Float16* v16 = k16 + (size_t)Cq * N;
  _Float16* P   = v16 + (size_t)C * N;

  float* out = (float*)d_out;

  for (int s = 0; s < 2; ++s) {
    const float* qw = (const float*)d_in[2 + 6 * s];
    const float* qb = (const float*)d_in[3 + 6 * s];
    const float* kw = (const float*)d_in[4 + 6 * s];
    const float* kb = (const float*)d_in[5 + 6 * s];
    const float* vw = (const float*)d_in[6 + 6 * s];
    const float* vb = (const float*)d_in[7 + 6 * s];

    for (int b = 0; b < B; ++b) {
      const float* x = inp[s] + (size_t)b * C * N;
      float*       o = out + (size_t)s * B * C * N + (size_t)b * C * N;

      proj_gemm_f16out<<<dim3(N / 64, Cq / 16), 32, 0, stream>>>(qw, qb, x, q16, C, N);
      proj_gemm_f16out<<<dim3(N / 64, Cq / 16), 32, 0, stream>>>(kw, kb, x, k16, C, N);
      proj_gemm_f16out<<<dim3(N / 64, C / 16),  32, 0, stream>>>(vw, vb, x, v16, C, N);

      attn_logits<<<dim3(N / 64, N / 16), 32, 0, stream>>>(q16, k16, P, Cq, N);
      softmax_rows<<<N, 256, 0, stream>>>(P, N);
      out_gemm_lds<<<dim3(N / OT_TN, C / OT_TM), 128, 0, stream>>>(v16, P, x, gamma, o, N);
    }
  }
}
